// DA_Rank_List_Proxy_Anchor_73194832658584
// MI455X (gfx1250) — compile-verified, matrix operand order checked by reading, not run on hardware
//
#include <hip/hip_runtime.h>
#include <hip/hip_bf16.h>

// ---------------- constants (match reference) ----------------
#define NB    4096      // batch
#define NC    10000     // classes / proxies
#define ND    512       // X / proxy dim
#define NDF   2048      // Feature dim

typedef __attribute__((ext_vector_type(16))) _Float16 v16h;
typedef __attribute__((ext_vector_type(8)))  _Float16 v8h;
typedef __attribute__((ext_vector_type(8)))  float    v8f;

// ---------------- helpers ----------------
__device__ __forceinline__ float blockReduceSum(float v) {
    __shared__ float red[1024];
    int tid = threadIdx.x;
    red[tid] = v;
    __syncthreads();
    for (int s = (int)blockDim.x >> 1; s > 0; s >>= 1) {
        if (tid < s) red[tid] += red[tid + s];
        __syncthreads();
    }
    float r = red[0];
    __syncthreads();
    return r;   // broadcast to all lanes
}

__device__ __forceinline__ v16h load_operand(const _Float16* rowBase, int k0, int h) {
    // ISA 16-bit operand layout: lane m+16h holds K = 8h..8h+7 (vgpr0-3)
    // and K = 16+8h..23+8h (vgpr4-7) of its row.
    const _Float16* p = rowBase + k0 + 8 * h;
    v8h lo = *reinterpret_cast<const v8h*>(p);
    v8h hi = *reinterpret_cast<const v8h*>(p + 16);
    return __builtin_shufflevector(lo, hi,
                0,1,2,3,4,5,6,7,8,9,10,11,12,13,14,15);
}

// ---------------- 0: zero the accumulator region of ws ----------------
__global__ void zero_ws_kernel(unsigned int* p, int ndw) {
    int i = blockIdx.x * blockDim.x + threadIdx.x;
    if (i < ndw) p[i] = 0u;
}

// ---------------- 1: per-row inverse L2 norm (x / sqrt(ss + 1e-12)) ----
__global__ __launch_bounds__(256) void rownorm_kernel(const float* __restrict__ x,
                                                      float* __restrict__ inv,
                                                      int cols) {
    int row = blockIdx.x;
    const float* p = x + (size_t)row * cols;
    float s = 0.f;
    for (int d = threadIdx.x; d < cols; d += 256) { float v = p[d]; s += v * v; }
    float tot = blockReduceSum(s);
    if (threadIdx.x == 0) inv[row] = rsqrtf(tot + 1e-12f);
}

// ---------------- 2: convert normalized rows to f16 --------------------
__global__ __launch_bounds__(256) void convert_f16_kernel(const float* __restrict__ x,
                                                          const float* __restrict__ inv,
                                                          _Float16* __restrict__ out,
                                                          int cols) {
    int row = blockIdx.x;
    float iv = inv[row];
    const float* p = x + (size_t)row * cols;
    _Float16* o = out + (size_t)row * cols;
    for (int d = threadIdx.x; d < cols; d += 256) o[d] = (_Float16)(p[d] * iv);
}

// ---------------- 3: class histogram -----------------------------------
__global__ void count_kernel(const int* __restrict__ T, int* __restrict__ counts) {
    int i = blockIdx.x * blockDim.x + threadIdx.x;
    if (i < NB) atomicAdd(&counts[T[i]], 1);
}

// ---------------- 4: single-block scan -> group tables ------------------
// Present classes in ascending id order get ranks 0..cnum-1 (== jnp.sort + cumsum).
__global__ __launch_bounds__(1024) void scan_groups_kernel(const int* __restrict__ counts,
                                                           int* __restrict__ gClass,
                                                           int* __restrict__ gOffset,
                                                           int* __restrict__ gCount,
                                                           int* __restrict__ nGroupsOut) {
    __shared__ int sCnt[1024], sPres[1024];
    __shared__ int carryCnt, carryPres;
    int tid = threadIdx.x;
    if (tid == 0) { carryCnt = 0; carryPres = 0; }
    __syncthreads();
    for (int base = 0; base < NC; base += 1024) {
        int c = base + tid;
        int cnt  = (c < NC) ? counts[c] : 0;
        int pres = (cnt > 0) ? 1 : 0;
        sCnt[tid] = cnt; sPres[tid] = pres;
        __syncthreads();
        // Hillis-Steele inclusive scan
        for (int off = 1; off < 1024; off <<= 1) {
            int a = 0, b = 0;
            if (tid >= off) { a = sCnt[tid - off]; b = sPres[tid - off]; }
            __syncthreads();
            sCnt[tid] += a; sPres[tid] += b;
            __syncthreads();
        }
        int incC = sCnt[tid], incP = sPres[tid];
        if (pres && c < NC) {
            int r = carryPres + incP - 1;        // group rank
            gClass[r]  = c;
            gOffset[r] = carryCnt + incC - cnt;  // start position in sorted order
            gCount[r]  = cnt;
        }
        __syncthreads();
        if (tid == 1023) { carryCnt += incC; carryPres += incP; }
        __syncthreads();
    }
    if (tid == 0) *nGroupsOut = carryPres;
}

// ---------------- 5: WMMA GEMM + fused epilogue --------------------------
// cos = Xn(f16) @ Pn(f16)^T ; per-column: Wsum[j] += exp(20*(0.4+cos))*[T!=j]
//                             Nnum[j] += w * relu(0.4+cos)
// Each wave: one 32(M) x 80(N) strip (2 M-tiles x 5 N-tiles = 10 C-tiles).
// 128 M-pairs x 125 N-strips = 16000 waves; B operand reused by 2 WMMAs.
__global__ __launch_bounds__(128) void gemm_epilogue_kernel(const _Float16* __restrict__ Xh,
                                                            const _Float16* __restrict__ Ph,
                                                            const int* __restrict__ T,
                                                            float* __restrict__ Wsum,
                                                            float* __restrict__ Nnum) {
    const int lane = threadIdx.x & 31;
    const int wave = threadIdx.x >> 5;
    int wid = blockIdx.x * 4 + wave;          // 0 .. 15999
    int mPair  = wid & 127;                   // 128 M blocks of 32 rows
    int nSuper = wid >> 7;                    // 125 N strips of 80
    if (nSuper >= 125) return;

    const int m = lane & 15;                  // row within tile (A) / col (B lane)
    const int h = lane >> 4;                  // half selector for K swizzle

    v8f acc[10];                              // [0..4]: M-tile0, [5..9]: M-tile1
#pragma unroll
    for (int t = 0; t < 10; ++t) acc[t] = (v8f){0.f,0.f,0.f,0.f,0.f,0.f,0.f,0.f};

    const _Float16* aRow0 = Xh + (size_t)(mPair * 32      + m) * ND;
    const _Float16* aRow1 = Xh + (size_t)(mPair * 32 + 16 + m) * ND;
    const _Float16* bRowBase = Ph + (size_t)(nSuper * 80 + m) * ND;

    for (int k0 = 0; k0 < ND; k0 += 32) {
        v16h A0 = load_operand(aRow0, k0, h);
        v16h A1 = load_operand(aRow1, k0, h);
#pragma unroll
        for (int t = 0; t < 5; ++t) {
            v16h Bv = load_operand(bRowBase + (size_t)(t * 16) * ND, k0, h);
            acc[t]     = __builtin_amdgcn_wmma_f32_16x16x32_f16(
                             false, A0, false, Bv, (short)0, acc[t],     false, false);
            acc[t + 5] = __builtin_amdgcn_wmma_f32_16x16x32_f16(
                             false, A1, false, Bv, (short)0, acc[t + 5], false, false);
        }
    }

    // epilogue: C tile layout -> lane l, vgpr v: row = v + 8*(l>>4), col = l&15
    int T0[8], T1[8];
#pragma unroll
    for (int v = 0; v < 8; ++v) {
        T0[v] = T[mPair * 32      + v + 8 * h];
        T1[v] = T[mPair * 32 + 16 + v + 8 * h];
    }

#pragma unroll
    for (int t = 0; t < 5; ++t) {
        int col = nSuper * 80 + t * 16 + (lane & 15);
        float sw = 0.f, sn = 0.f;
#pragma unroll
        for (int v = 0; v < 8; ++v) {
            float c = acc[t][v];
            float r = 0.4f + c;                  // m - cos_dis
            float w = __expf(20.0f * r);         // exp(t*(m - cos_dis))
            if (T0[v] == col) w = 0.f;           // N1 mask
            sw += w;
            sn += w * fmaxf(r, 0.f);

            float c2 = acc[t + 5][v];
            float r2 = 0.4f + c2;
            float w2 = __expf(20.0f * r2);
            if (T1[v] == col) w2 = 0.f;
            sw += w2;
            sn += w2 * fmaxf(r2, 0.f);
        }
        atomicAdd(&Wsum[col], sw);
        atomicAdd(&Nnum[col], sn);
    }
}

// ---------------- 6: positive term (diagonal dots, full fp32) ------------
__global__ __launch_bounds__(256) void pos_kernel(const float* __restrict__ X,
                                                  const float* __restrict__ P,
                                                  const int* __restrict__ T,
                                                  const float* __restrict__ invX,
                                                  const float* __restrict__ invP,
                                                  float* __restrict__ posSum) {
    int i = blockIdx.x;
    int cls = T[i];
    const float* xp = X + (size_t)i * ND;
    const float* pp = P + (size_t)cls * ND;
    float s = 0.f;
    for (int d = threadIdx.x; d < ND; d += 256) s += xp[d] * pp[d];
    float dot = blockReduceSum(s);
    if (threadIdx.x == 0) {
        float cosv = dot * invX[i] * invP[cls];
        // pos_exp = relu(cos_dis - (m-mrg)) = relu(-cos)
        atomicAdd(posSum, fmaxf(-cosv, 0.f));
    }
}

// ---------------- 7: per-group DA branch ---------------------------------
// acc: [0]=num1  [1]=num2  [2]=S_e2  [3]=S_ea  [4]=S_a2
__global__ __launch_bounds__(256) void group_kernel(const float* __restrict__ X,
                                                    const float* __restrict__ Feature,
                                                    const float* __restrict__ P,
                                                    const float* __restrict__ alphac,
                                                    const float* __restrict__ invX,
                                                    const float* __restrict__ invF,
                                                    const float* __restrict__ invP,
                                                    const int* __restrict__ gClass,
                                                    const int* __restrict__ gOffset,
                                                    const int* __restrict__ gCount,
                                                    const int* __restrict__ nGroups,
                                                    float* __restrict__ acc) {
    int k = blockIdx.x;
    if (k >= *nGroups) return;
    int cls = gClass[k], off = gOffset[k], cnt = gCount[k];
    float a = alphac[cls];
    int tid = threadIdx.x;

    __shared__ float us[NDF];                 // per-group Feature sum (8 KB)
    __shared__ float sDX, sD0;
    if (tid == 0) { sDX = 0.f; sD0 = 0.f; }

    for (int d = tid; d < NDF; d += 256) {
        float s = 0.f;
        for (int i = 0; i < cnt; ++i) s += Feature[(size_t)(off + i) * NDF + d];
        us[d] = s;
    }
    __syncthreads();

    float s2 = 0.f;
    for (int d = tid; d < NDF; d += 256) { float v = us[d]; s2 += v * v; }
    float invU = rsqrtf(blockReduceSum(s2) + 1e-12f);

    const float ivp = invP[k];                // NOTE: proxy row by GROUP RANK (ref semantics)
    for (int i = 0; i < cnt; ++i) {
        int row = off + i;                    // original-order rows at sorted positions
        float ivx = invX[row];
        float s = 0.f;
        for (int d = tid; d < ND; d += 256) {
            float diff = X[(size_t)row * ND + d] * ivx
                       - P[(size_t)k   * ND + d] * ivp + 1e-6f;
            s += diff * diff;
        }
        float tot = blockReduceSum(s);
        if (tid == 0) sDX += sqrtf(tot);

        float ivf = invF[row];
        s = 0.f;
        for (int d = tid; d < NDF; d += 256) {
            float diff = Feature[(size_t)row * NDF + d] * ivf - us[d] * invU + 1e-6f;
            s += diff * diff;
        }
        tot = blockReduceSum(s);
        if (tid == 0) sD0 += sqrtf(tot);
    }
    __syncthreads();
    if (tid == 0) {
        float rc = 1.0f / (float)cnt;
        float Davg  = sDX * rc;
        float davg0 = sD0 * rc;               // e = sqrt(davg0), e^2 = davg0
        atomicAdd(&acc[0], (Davg - a) * (Davg - a));
        atomicAdd(&acc[1], a);
        atomicAdd(&acc[2], davg0);
        atomicAdd(&acc[3], sqrtf(davg0) * a);
        atomicAdd(&acc[4], a * a);
    }
}

// ---------------- 8: final combine ---------------------------------------
__global__ __launch_bounds__(1024) void final_kernel(const float* __restrict__ Wsum,
                                                     const float* __restrict__ Nnum,
                                                     const float* __restrict__ posSum,
                                                     const float* __restrict__ acc,
                                                     const int* __restrict__ nGroups,
                                                     float* __restrict__ out) {
    float s = 0.f;
    for (int j = threadIdx.x; j < NC; j += 1024) {
        float w = Wsum[j];
        s += (w > 0.f) ? (Nnum[j] / w) : 0.f;
    }
    float negSum = blockReduceSum(s);
    if (threadIdx.x == 0) {
        float cnum = (float)(*nGroups);
        float nb   = (float)NC;
        float pos_term = posSum[0] / cnum;    // num_valid == cnum
        float neg_term = negSum / nb;
        float num1 = acc[0], num2 = acc[1];
        float Se2 = acc[2], Sea = acc[3], Sa2 = acc[4];
        float inter = (Sa2 * Se2 - Sea * Sea) / (cnum * cnum);
        float LDA = num1 / nb - num2 / nb + inter;
        out[0] = pos_term + neg_term + 10.0f * LDA;
    }
}

// ---------------- launch -------------------------------------------------
extern "C" void kernel_launch(void* const* d_in, const int* in_sizes, int n_in,
                              void* d_out, int out_size, void* d_ws, size_t ws_size,
                              hipStream_t stream) {
    const float* X       = (const float*)d_in[0];
    const int*   T       = (const int*)  d_in[1];
    const float* Feature = (const float*)d_in[2];
    const float* P       = (const float*)d_in[3];
    const float* alphac  = (const float*)d_in[4];
    float* out = (float*)d_out;

    // workspace layout
    char* ws = (char*)d_ws;
    size_t o = 0;
    auto take = [&](size_t bytes, size_t align) -> char* {
        o = (o + align - 1) & ~(align - 1);
        char* p = ws + o; o += bytes; return p;
    };
    float* Wsum    = (float*)take(NC * 4, 4);
    float* Nnum    = (float*)take(NC * 4, 4);
    int*   counts  = (int*)  take(NC * 4, 4);
    int*   gClass  = (int*)  take(NB * 4, 4);
    int*   gOffset = (int*)  take(NB * 4, 4);
    int*   gCount  = (int*)  take(NB * 4, 4);
    int*   nGroups = (int*)  take(4, 4);
    float* posSum  = (float*)take(4, 4);
    float* acc5    = (float*)take(5 * 4, 4);
    size_t zbytes  = o;                        // zero everything above each call
    float* invX    = (float*)take(NB * 4, 256);
    float* invF    = (float*)take(NB * 4, 4);
    float* invP    = (float*)take(NC * 4, 4);
    _Float16* Xh   = (_Float16*)take((size_t)NB * ND * 2, 256);
    _Float16* Ph   = (_Float16*)take((size_t)NC * ND * 2, 256);
    (void)ws_size; (void)in_sizes; (void)n_in; (void)out_size;

    // 0) zero accumulators
    int ndw = (int)(zbytes / 4);
    zero_ws_kernel<<<(ndw + 255) / 256, 256, 0, stream>>>((unsigned int*)ws, ndw);

    // 1) row inverse norms
    rownorm_kernel<<<NB, 256, 0, stream>>>(X, invX, ND);
    rownorm_kernel<<<NB, 256, 0, stream>>>(Feature, invF, NDF);
    rownorm_kernel<<<NC, 256, 0, stream>>>(P, invP, ND);

    // 2) f16 normalized copies for WMMA
    convert_f16_kernel<<<NB, 256, 0, stream>>>(X, invX, Xh, ND);
    convert_f16_kernel<<<NC, 256, 0, stream>>>(P, invP, Ph, ND);

    // 3) class histogram + group tables
    count_kernel<<<(NB + 255) / 256, 256, 0, stream>>>(T, counts);
    scan_groups_kernel<<<1, 1024, 0, stream>>>(counts, gClass, gOffset, gCount, nGroups);

    // 4) WMMA GEMM + fused negative-term epilogue
    //    128 M-pairs x 125 N-strips = 16000 waves, 4 waves/block
    gemm_epilogue_kernel<<<4000, 128, 0, stream>>>(Xh, Ph, T, Wsum, Nnum);

    // 5) positive term
    pos_kernel<<<NB, 256, 0, stream>>>(X, P, T, invX, invP, posSum);

    // 6) DA branch per group
    group_kernel<<<NB, 256, 0, stream>>>(X, Feature, P, alphac, invX, invF, invP,
                                         gClass, gOffset, gCount, nGroups, acc5);

    // 7) combine
    final_kernel<<<1, 1024, 0, stream>>>(Wsum, Nnum, posSum, acc5, nGroups, out);
}
